// ETTRegressionLayer_LSPUF_23038204576032
// MI455X (gfx1250) — matrix operand order repaired and checked by reading, not compiled, not run on hardware
//
#include <hip/hip_runtime.h>
#include <hip/hip_bf16.h>

typedef __attribute__((ext_vector_type(16))) __bf16 v16bf;
typedef __attribute__((ext_vector_type(8)))  float  v8f;

#define PHI      65
#define XLD      (8 * PHI)   // 520, input row stride
#define MTILE    128
#define KSTR     72          // bf16 k-stride for weight LDS (banking)
#define TOTCOLP  416         // 16 + 6*64 + 16 padded columns
#define RES_LD   401         // f32 result row stride (odd -> conflict-free)

union ABfrag { v16bf v; unsigned int u[8]; };

__device__ __forceinline__ unsigned short bf16_rne(float x) {
  unsigned int u = __float_as_uint(x);
  unsigned int r = u + 0x7fffu + ((u >> 16) & 1u);
  return (unsigned short)(r >> 16);
}

// padded weight-column base per core c (0..7)
__device__ __forceinline__ constexpr int CB(int c)    { return (c == 0) ? 0 : (c < 7 ? 16 + (c - 1) * 64 : 400); }
// result-column base per core c
__device__ __forceinline__ constexpr int RCB(int c)   { return (c == 0) ? 0 : (c < 7 ? 8 + (c - 1) * 64 : 392); }
__device__ __forceinline__ constexpr int NTIL(int c)  { return (c == 0 || c == 7) ? 1 : 4; }
__device__ __forceinline__ constexpr int NREAL(int c) { return (c == 0 || c == 7) ? 8 : 64; }

__global__ __launch_bounds__(256) void tt_regression_kernel(
    const float* __restrict__ inputs,     // (B, 520)
    const float* __restrict__ core0,      // (1, 8, 65)
    const float* __restrict__ cores_mid,  // (6, 8, 8, 65)
    const float* __restrict__ core_last,  // (8, 1, 65)
    const float* __restrict__ out_tensor, // (1,1)
    const float* __restrict__ bias,       // (1,)
    float* __restrict__ out)              // (B, 1)
{
  __shared__ unsigned short wtr[TOTCOLP * KSTR]; // bf16 weights, [col][k], k=0..63 valid
  __shared__ float w64s[TOTCOLP];                // f32 k=64 weight row
  __shared__ float res[MTILE][RES_LD];           // stage-1 results

  const int tid = threadIdx.x;

  // ---------------- stage 0: weights -> LDS (bf16, transposed) ----------------
  for (int idx = tid; idx < TOTCOLP * 64; idx += 256) {
    const int col = idx >> 6;
    const int k   = idx & 63;
    float w = 0.0f;
    if (col < 8)                       w = core0[col * PHI + k];
    else if (col >= 16 && col < 400)   w = cores_mid[(col - 16) * PHI + k]; // (c*64+p*8+q)*65+k
    else if (col >= 400 && col < 408)  w = core_last[(col - 400) * PHI + k];
    wtr[col * KSTR + k] = bf16_rne(w);
  }
  for (int col = tid; col < TOTCOLP; col += 256) {
    float w = 0.0f;
    if (col < 8)                       w = core0[col * PHI + 64];
    else if (col >= 16 && col < 400)   w = cores_mid[(col - 16) * PHI + 64];
    else if (col >= 400 && col < 408)  w = core_last[(col - 400) * PHI + 64];
    w64s[col] = w;
  }
  __syncthreads();

  // ---------------- stage 1: eight GEMMs via bf16 WMMA ----------------
  const int  wave = tid >> 5;
  const int  lane = tid & 31;
  const int  hf   = lane >> 4;   // lane half (K-split for A/B, M-split for C)
  const int  ln   = lane & 15;   // A: M row, B/C: N column
  const long m0   = (long)blockIdx.x * MTILE + wave * 16;

#pragma unroll
  for (int c = 0; c < 8; ++c) {
    // --- A fragments (kt=0,1) straight from global f32, packed to bf16 via v_perm ---
    const float* p0 = inputs + (m0 + ln) * XLD + c * PHI + hf * 8;
    ABfrag a0, a1;
    {
      float f[16];
#pragma unroll
      for (int i = 0; i < 8; ++i) { f[i] = p0[i]; f[8 + i] = p0[16 + i]; }
#pragma unroll
      for (int j = 0; j < 8; ++j)
        a0.u[j] = __builtin_amdgcn_perm(__float_as_uint(f[2 * j + 1]),
                                        __float_as_uint(f[2 * j]), 0x07060302u);
#pragma unroll
      for (int i = 0; i < 8; ++i) { f[i] = p0[32 + i]; f[8 + i] = p0[48 + i]; }
#pragma unroll
      for (int j = 0; j < 8; ++j)
        a1.u[j] = __builtin_amdgcn_perm(__float_as_uint(f[2 * j + 1]),
                                        __float_as_uint(f[2 * j]), 0x07060302u);
    }
    // k=64 activations for the exact f32 epilogue (row = hf*8 + i of this M-tile)
    float xk[8];
#pragma unroll
    for (int i = 0; i < 8; ++i)
      xk[i] = inputs[(m0 + hf * 8 + i) * XLD + c * PHI + 64];

#pragma unroll
    for (int nt = 0; nt < NTIL(c); ++nt) {
      const int col = nt * 16 + ln;
      // --- B fragments: two 16B runs per lane from LDS (k = 8h..8h+7, 16+8h..16+8h+7) ---
      const unsigned int* wp =
          (const unsigned int*)&wtr[(CB(c) + col) * KSTR + hf * 8];
      ABfrag b0, b1;
#pragma unroll
      for (int j = 0; j < 8; ++j) b0.u[j] = wp[j + (j >> 2) * 4];
#pragma unroll
      for (int j = 0; j < 8; ++j) b1.u[j] = wp[16 + j + (j >> 2) * 4]; // +32 bf16 = kt 1

      v8f acc = {0.f, 0.f, 0.f, 0.f, 0.f, 0.f, 0.f, 0.f};
      acc = __builtin_amdgcn_wmma_f32_16x16x32_bf16(false, a0.v, false, b0.v,
                                                    (short)0, acc, false, false);
      acc = __builtin_amdgcn_wmma_f32_16x16x32_bf16(false, a1.v, false, b1.v,
                                                    (short)0, acc, false, false);
      // rank-1 f32 epilogue for k=64
      const float wv = w64s[CB(c) + col];
#pragma unroll
      for (int i = 0; i < 8; ++i) acc[i] += xk[i] * wv;

      if (col < NREAL(c)) {
#pragma unroll
        for (int i = 0; i < 8; ++i)
          res[wave * 16 + hf * 8 + i][RCB(c) + col] = acc[i];
      }
    }
  }
  __syncthreads();

  // ---------------- stage 2: per-sample chain product ----------------
  if (tid < MTILE) {
    float v[8];
#pragma unroll
    for (int q = 0; q < 8; ++q) v[q] = res[tid][q];
#pragma unroll
    for (int c = 0; c < 6; ++c) {
      float nv[8];
#pragma unroll
      for (int q = 0; q < 8; ++q) {
        float s = 0.f;
#pragma unroll
        for (int p = 0; p < 8; ++p)
          s += v[p] * res[tid][8 + c * 64 + p * 8 + q];
        nv[q] = s;
      }
#pragma unroll
      for (int q = 0; q < 8; ++q) v[q] = nv[q];
    }
    float s = 0.f;
#pragma unroll
    for (int p = 0; p < 8; ++p) s += v[p] * res[tid][392 + p];
    out[(long)blockIdx.x * MTILE + tid] = s * out_tensor[0] + bias[0];
  }
}

extern "C" void kernel_launch(void* const* d_in, const int* in_sizes, int n_in,
                              void* d_out, int out_size, void* d_ws, size_t ws_size,
                              hipStream_t stream) {
  const float* inputs    = (const float*)d_in[0];
  const float* core0     = (const float*)d_in[1];
  const float* cores_mid = (const float*)d_in[2];
  const float* core_last = (const float*)d_in[3];
  const float* outt      = (const float*)d_in[4];
  const float* bias      = (const float*)d_in[5];
  float* out             = (float*)d_out;

  const int B = in_sizes[0] / XLD;  // 131072
  dim3 grid(B / MTILE), block(256);
  tt_regression_kernel<<<grid, block, 0, stream>>>(inputs, core0, cores_mid,
                                                   core_last, outt, bias, out);
}
